// GRU_16647293240150
// MI455X (gfx1250) — compile-verified, matrix operand order checked by reading, not run on hardware
//
#include <hip/hip_runtime.h>
#include <hip/hip_bf16.h>

#define B_ 32
#define T_ 2048
#define I_ 256
#define H_ 512
#define G_ 1536   // 3*H
#define C_ 64
#define NWG 8     // workgroups in recurrent kernel (each owns 64 h-columns)

typedef __attribute__((ext_vector_type(16))) _Float16 v16h;
typedef __attribute__((ext_vector_type(8)))  _Float16 v8h;
typedef __attribute__((ext_vector_type(8)))  float    v8f;

// ---- tile loaders -------------------------------------------------------
// 16-bit A/B WMMA operand layout (CDNA5 ISA 7.12.2): lane L (&15) = row,
// lane>>4 selects K-half; per lane: 8 contiguous f16 at +0 and 8 at +16.
__device__ __forceinline__ v16h load_tile(const _Float16* p) {
  v8h lo = *reinterpret_cast<const v8h*>(p);
  v8h hi = *reinterpret_cast<const v8h*>(p + 16);
  return __builtin_shufflevector(lo, hi, 0,1,2,3,4,5,6,7,8,9,10,11,12,13,14,15);
}

__device__ __forceinline__ float sigmoidf_(float x) {
  return 1.0f / (1.0f + __expf(-x));
}
__device__ __forceinline__ float tanhf_(float x) {
  x = fminf(15.0f, fmaxf(-15.0f, x));
  float e = __expf(2.0f * x);
  return (e - 1.0f) / (e + 1.0f);
}

// ---- f32 -> f16 convert -------------------------------------------------
__global__ void __launch_bounds__(256)
gru_f32_to_f16(const float* __restrict__ src, _Float16* __restrict__ dst, int n) {
  int i = blockIdx.x * 256 + threadIdx.x;
  if (i < n) dst[i] = (_Float16)src[i];
}

// ---- gx GEMM: out[m, n] = sum_k A[m,k] * W[n,k] + bias[n] ---------------
// A: [M x K] f16 row-major (lda = K). W: [1536 x K] f16 row-major.
// out: [M x 1536] f16. One wave per block computes a 16x128 output strip.
__global__ void __launch_bounds__(32)
gru_gx_gemm(const _Float16* __restrict__ A, const _Float16* __restrict__ W,
            const float* __restrict__ bias, _Float16* __restrict__ out, int K) {
  const int lane = threadIdx.x;
  const int mtile = blockIdx.x;          // M/16 tiles
  const int n0 = blockIdx.y * 128;       // 12 strips over N=1536
  const int row = lane & 15;
  const int koff = (lane >> 4) << 3;

  v8f acc[8];
#pragma unroll
  for (int j = 0; j < 8; ++j) acc[j] = (v8f){0.f,0.f,0.f,0.f,0.f,0.f,0.f,0.f};

  for (int k0 = 0; k0 < K; k0 += 32) {
    v16h a = load_tile(A + (size_t)(mtile * 16 + row) * K + k0 + koff);
#pragma unroll
    for (int j = 0; j < 8; ++j) {
      v16h b = load_tile(W + (size_t)(n0 + j * 16 + row) * K + k0 + koff);
      acc[j] = __builtin_amdgcn_wmma_f32_16x16x32_f16(
          false, a, false, b, (short)0, acc[j], false, false);
    }
  }
  // D layout: lane<16 owns M rows 0..7, lane>=16 owns 8..15; column = lane&15.
  const int m0 = mtile * 16 + ((lane >> 4) << 3);
#pragma unroll
  for (int j = 0; j < 8; ++j) {
    int n = n0 + j * 16 + row;
    float bi = bias[n];
#pragma unroll
    for (int i = 0; i < 8; ++i)
      out[(size_t)(m0 + i) * G_ + n] = (_Float16)(acc[j][i] + bi);
  }
}

// ---- device-wide barrier (state memset to 0 before launch) --------------
__device__ __forceinline__ void grid_barrier(unsigned* cnt, unsigned* gen,
                                             unsigned target) {
  __threadfence();
  __syncthreads();
  if (threadIdx.x == 0) {
    unsigned old = atomicAdd(cnt, 1u);
    if (old == gridDim.x - 1) {
      atomicExch(cnt, 0u);
      atomicAdd(gen, 1u);
    }
    while (atomicAdd(gen, 0u) < target) {
      __builtin_amdgcn_s_sleep(1);
    }
  }
  __syncthreads();
  __threadfence();
}

// ---- persistent GRU recurrence -----------------------------------------
// NWG blocks of 256 threads (8 waves). Block wg owns h-columns [wg*64, wg*64+64).
// Its W_hh slice (r,z,n rows for those columns) lives in LDS: [192][512] f16.
// Wave w: mtile = w&1 (batch rows 0-15 / 16-31), col-slice cs = w>>1 (16 cols).
// Per step: gh = h @ W_slice^T via 16 K-steps x 3 gates of wmma_f32_16x16x32_f16,
// gate math, write h_next, grid barrier, ping-pong swap. gx for step t+1 is
// prefetched (global_prefetch_b8) while step t computes.
__global__ void __launch_bounds__(256)
gru_recur(const _Float16* __restrict__ gx,  // [B*T, 1536], bt = b*T + t
          const _Float16* __restrict__ Wg,  // [1536, 512] f16
          const float* __restrict__ b_hh,   // [1536]
          _Float16* h_a,                    // ping (zero-initialized)
          _Float16* h_b,                    // pong (zero-initialized)
          _Float16* hseq,                   // [B*T, 512] f16 or nullptr
          unsigned* bar_cnt, unsigned* bar_gen) {
  extern __shared__ _Float16 Wl[];  // 192 rows x 512 cols
  const int tid = threadIdx.x;
  const int wg = blockIdx.x;
  const int colwg = wg * 64;

  // Preload weight slice into LDS (v8h chunks).
  for (int idx = tid; idx < (192 * 512) / 8; idx += 256) {
    int flat = idx * 8;
    int lrow = flat >> 9;          // local row 0..191
    int k = flat & 511;
    int gate = lrow >> 6;          // 0=r,1=z,2=n
    int grow = gate * H_ + colwg + (lrow & 63);
    *reinterpret_cast<v8h*>(&Wl[(size_t)lrow * H_ + k]) =
        *reinterpret_cast<const v8h*>(&Wg[(size_t)grow * H_ + k]);
  }
  __syncthreads();

  const int wave = tid >> 5, lane = tid & 31;
  const int mtile = wave & 1;
  const int cs = wave >> 1;                      // 0..3
  const int row = lane & 15;
  const int koff = (lane >> 4) << 3;
  const int arow = mtile * 16 + row;             // A (h) row for this lane
  const int m0 = mtile * 16 + ((lane >> 4) << 3);
  const int col = colwg + cs * 16 + row;         // owned h column
  const float bhr = b_hh[col];
  const float bhz = b_hh[H_ + col];
  const float bhn = b_hh[2 * H_ + col];
  const size_t lds_r = (size_t)(cs * 16 + row) * H_;
  const size_t lds_z = (size_t)(64 + cs * 16 + row) * H_;
  const size_t lds_n = (size_t)(128 + cs * 16 + row) * H_;

  const _Float16* hcur = h_a;
  _Float16* hnxt = h_b;

  // Warm the gx stream for t=0.
#pragma unroll
  for (int i = 0; i < 8; ++i) {
    const _Float16* gxp = gx + ((size_t)(m0 + i) * T_) * G_;
    __builtin_prefetch(gxp + col, 0, 1);
    __builtin_prefetch(gxp + H_ + col, 0, 1);
    __builtin_prefetch(gxp + 2 * H_ + col, 0, 1);
  }

#pragma unroll 1
  for (int t = 0; t < T_; ++t) {
    v8f ar = (v8f){0.f,0.f,0.f,0.f,0.f,0.f,0.f,0.f};
    v8f az = ar, an = ar;
#pragma unroll 4
    for (int k0 = 0; k0 < H_; k0 += 32) {
      v16h a = load_tile(hcur + (size_t)arow * H_ + k0 + koff);
      v16h br = load_tile(&Wl[lds_r + k0 + koff]);
      ar = __builtin_amdgcn_wmma_f32_16x16x32_f16(false, a, false, br,
                                                  (short)0, ar, false, false);
      v16h bz = load_tile(&Wl[lds_z + k0 + koff]);
      az = __builtin_amdgcn_wmma_f32_16x16x32_f16(false, a, false, bz,
                                                  (short)0, az, false, false);
      v16h bn = load_tile(&Wl[lds_n + k0 + koff]);
      an = __builtin_amdgcn_wmma_f32_16x16x32_f16(false, a, false, bn,
                                                  (short)0, an, false, false);
    }
    // Prefetch next timestep's gx lines while this step's epilogue runs.
    if (t + 1 < T_) {
#pragma unroll
      for (int i = 0; i < 8; ++i) {
        const _Float16* gxn = gx + ((size_t)(m0 + i) * T_ + (t + 1)) * G_;
        __builtin_prefetch(gxn + col, 0, 1);
        __builtin_prefetch(gxn + H_ + col, 0, 1);
        __builtin_prefetch(gxn + 2 * H_ + col, 0, 1);
      }
    }
#pragma unroll
    for (int i = 0; i < 8; ++i) {
      int m = m0 + i;  // batch index
      const _Float16* gxp = gx + ((size_t)m * T_ + t) * G_;
      float gr = (float)gxp[col];
      float gz = (float)gxp[H_ + col];
      float gn = (float)gxp[2 * H_ + col];
      float r = sigmoidf_(gr + ar[i] + bhr);
      float z = sigmoidf_(gz + az[i] + bhz);
      float nn = tanhf_(gn + r * (an[i] + bhn));
      float hold = (float)hcur[(size_t)m * H_ + col];
      float hnew = (1.0f - z) * nn + z * hold;
      hnxt[(size_t)m * H_ + col] = (_Float16)hnew;
      if (hseq) hseq[((size_t)m * T_ + t) * H_ + col] = (_Float16)hnew;
    }
    grid_barrier(bar_cnt, bar_gen, (unsigned)(t + 1));
    const _Float16* tmp = hcur;
    hcur = hnxt;
    hnxt = (_Float16*)tmp;
  }
}

// ---- FC + softmax -------------------------------------------------------
__global__ void __launch_bounds__(64)
gru_fc_softmax(const _Float16* __restrict__ h, const float* __restrict__ fc_w,
               const float* __restrict__ fc_b, float* __restrict__ out) {
  int b = blockIdx.x, c = threadIdx.x;
  float acc = fc_b[c];
  for (int k = 0; k < H_; ++k)
    acc += (float)h[(size_t)b * H_ + k] * fc_w[(size_t)c * H_ + k];
  __shared__ float sl[C_];
  sl[c] = acc;
  __syncthreads();
  float mx = sl[0];
  for (int k = 1; k < C_; ++k) mx = fmaxf(mx, sl[k]);
  float e = __expf(acc - mx);
  __shared__ float se[C_];
  se[c] = e;
  __syncthreads();
  float s = 0.f;
  for (int k = 0; k < C_; ++k) s += se[k];
  out[b * C_ + c] = e / s;
}

// ---- host side ----------------------------------------------------------
extern "C" void kernel_launch(void* const* d_in, const int* in_sizes, int n_in,
                              void* d_out, int out_size, void* d_ws, size_t ws_size,
                              hipStream_t stream) {
  const float* x      = (const float*)d_in[0];
  const float* w_ih_0 = (const float*)d_in[1];
  const float* w_hh_0 = (const float*)d_in[2];
  const float* b_ih_0 = (const float*)d_in[3];
  const float* b_hh_0 = (const float*)d_in[4];
  const float* w_ih_1 = (const float*)d_in[5];
  const float* w_hh_1 = (const float*)d_in[6];
  const float* b_ih_1 = (const float*)d_in[7];
  const float* b_hh_1 = (const float*)d_in[8];
  const float* fc_w   = (const float*)d_in[9];
  const float* fc_b   = (const float*)d_in[10];
  float* out = (float*)d_out;

  char* ws = (char*)d_ws;
  size_t off = 0;
  auto alloc = [&](size_t bytes) -> void* {
    void* p = ws + off;
    off += (bytes + 255) & ~(size_t)255;
    return p;
  };
  _Float16* xh    = (_Float16*)alloc((size_t)B_ * T_ * I_ * 2);
  _Float16* wih0h = (_Float16*)alloc((size_t)G_ * I_ * 2);
  _Float16* whh0h = (_Float16*)alloc((size_t)G_ * H_ * 2);
  _Float16* wih1h = (_Float16*)alloc((size_t)G_ * H_ * 2);
  _Float16* whh1h = (_Float16*)alloc((size_t)G_ * H_ * 2);
  _Float16* gxbuf = (_Float16*)alloc((size_t)B_ * T_ * G_ * 2);
  _Float16* hseq  = (_Float16*)alloc((size_t)B_ * T_ * H_ * 2);
  _Float16* hping = (_Float16*)alloc((size_t)B_ * H_ * 2);
  _Float16* hpong = (_Float16*)alloc((size_t)B_ * H_ * 2);
  unsigned* bar   = (unsigned*)alloc(256);
  unsigned* bar_cnt = bar;
  unsigned* bar_gen = bar + 1;
  const size_t zero_span = (size_t)B_ * H_ * 2 * 2 + 256;  // hping+hpong+bar

  // f32 -> f16 converts
  {
    int n = B_ * T_ * I_;
    gru_f32_to_f16<<<(n + 255) / 256, 256, 0, stream>>>(x, xh, n);
    n = G_ * I_;
    gru_f32_to_f16<<<(n + 255) / 256, 256, 0, stream>>>(w_ih_0, wih0h, n);
    n = G_ * H_;
    gru_f32_to_f16<<<(n + 255) / 256, 256, 0, stream>>>(w_hh_0, whh0h, n);
    gru_f32_to_f16<<<(n + 255) / 256, 256, 0, stream>>>(w_ih_1, wih1h, n);
    gru_f32_to_f16<<<(n + 255) / 256, 256, 0, stream>>>(w_hh_1, whh1h, n);
  }

  const int MT = (B_ * T_) / 16;  // 4096 M-tiles
  const size_t lds_bytes = (size_t)192 * H_ * 2;  // 196608 B per WGP

  // ---- layer 0 ----
  gru_gx_gemm<<<dim3(MT, G_ / 128), 32, 0, stream>>>(xh, wih0h, b_ih_0, gxbuf, I_);
  hipMemsetAsync(hping, 0, zero_span, stream);
  gru_recur<<<NWG, 256, lds_bytes, stream>>>(gxbuf, whh0h, b_hh_0, hping, hpong,
                                             hseq, bar_cnt, bar_gen);
  // ---- layer 1 ----
  gru_gx_gemm<<<dim3(MT, G_ / 128), 32, 0, stream>>>(hseq, wih1h, b_ih_1, gxbuf, H_);
  hipMemsetAsync(hping, 0, zero_span, stream);
  gru_recur<<<NWG, 256, lds_bytes, stream>>>(gxbuf, whh1h, b_hh_1, hping, hpong,
                                             nullptr, bar_cnt, bar_gen);

  // T=2048 steps: last step (t=2047, odd) wrote hping.
  gru_fc_softmax<<<B_, C_, 0, stream>>>(hping, fc_w, fc_b, out);
}
